// WSDModel_1967095022320
// MI455X (gfx1250) — matrix operand order repaired and checked by reading, not compiled, not run on hardware
//
#include <hip/hip_runtime.h>

// ---------------- problem dims ----------------
#define D_DIM   1024
#define B_DIM   4
#define N_CTX   2048
#define V_VOC   50257
#define Y_VOC   20000
#define ROWS    (B_DIM * N_CTX)          // 8192
#define NEG_INF (-1e9f)
#define LN_EPS  (1e-5f)

// ---------------- CDNA5 async-to-LDS support ----------------
#if defined(__AMDGCN__) && __has_builtin(__builtin_amdgcn_global_load_async_to_lds_b128)
#define HAS_ASYNC 1
typedef int v4i_async __attribute__((vector_size(16)));
typedef __attribute__((address_space(1))) v4i_async* as1_v4i;
typedef __attribute__((address_space(3))) v4i_async* as3_v4i;
#endif

__device__ __forceinline__ void async_copy16(const unsigned short* g, unsigned short* l) {
#ifdef HAS_ASYNC
    __builtin_amdgcn_global_load_async_to_lds_b128((as1_v4i)g, (as3_v4i)l, 0, 0);
#else
    (void)g; (void)l;
#endif
}

template<int N_OUT>
__device__ __forceinline__ void wait_async() {
#ifdef HAS_ASYNC
#if __has_builtin(__builtin_amdgcn_s_wait_asynccnt)
    __builtin_amdgcn_s_wait_asynccnt(N_OUT);
#else
    asm volatile("s_wait_asynccnt %0" :: "n"(N_OUT) : "memory");
#endif
#endif
}

// ---------------- WMMA types ------------------
typedef __attribute__((ext_vector_type(16))) __bf16 v16bf;
typedef __attribute__((ext_vector_type(8)))  float  v8f;

union Frag16 { v16bf v; uint4 q[2]; };

__device__ __forceinline__ unsigned short f2bf(float x) {
    unsigned int u = __float_as_uint(x);
    unsigned int r = u + 0x7FFFu + ((u >> 16) & 1u);   // round-to-nearest-even
    return (unsigned short)(r >> 16);
}

// ---------------- helper kernels ---------------
__global__ __launch_bounds__(256) void f32_to_bf16_kernel(
    const float* __restrict__ src, unsigned short* __restrict__ dst, size_t n) {
    size_t i = (size_t)blockIdx.x * 256 + threadIdx.x;
    size_t stride = (size_t)gridDim.x * 256;
    for (; i < n; i += stride) dst[i] = f2bf(src[i]);
}

__global__ __launch_bounds__(256) void gather_embed_kernel(
    const int* __restrict__ Ms, const float* __restrict__ Ev,
    float* __restrict__ Xf, unsigned short* __restrict__ Xbf) {
    int row = blockIdx.x;               // 0..8191
    int tok = Ms[row];
    const float* src = Ev + (size_t)tok * D_DIM;
    size_t base = (size_t)row * D_DIM;
    for (int i = threadIdx.x; i < D_DIM; i += 256) {
        float v = src[i];
        Xf[base + i]  = v;
        Xbf[base + i] = f2bf(v);
    }
}

// row softmax over N_CTX elements; writes A (f32, to d_out tail) and bf16 copy
__global__ __launch_bounds__(256) void softmax_kernel(
    const float* __restrict__ L, float* __restrict__ Aout,
    unsigned short* __restrict__ Abf) {
    __shared__ float red[256];
    int row = blockIdx.x;               // 0..8191  (= b*2048 + q)
    int t = threadIdx.x;
    const float* lr = L + (size_t)row * N_CTX;
    float v[8];
    float m = -3.0e38f;
    #pragma unroll
    for (int i = 0; i < 8; ++i) { v[i] = lr[t + i * 256]; m = fmaxf(m, v[i]); }
    red[t] = m; __syncthreads();
    for (int s = 128; s > 0; s >>= 1) { if (t < s) red[t] = fmaxf(red[t], red[t + s]); __syncthreads(); }
    m = red[0]; __syncthreads();
    float sum = 0.f;
    #pragma unroll
    for (int i = 0; i < 8; ++i) { v[i] = __expf(v[i] - m); sum += v[i]; }
    red[t] = sum; __syncthreads();
    for (int s = 128; s > 0; s >>= 1) { if (t < s) red[t] += red[t + s]; __syncthreads(); }
    float inv = 1.0f / red[0];
    size_t base = (size_t)row * N_CTX;
    #pragma unroll
    for (int i = 0; i < 8; ++i) {
        float a = v[i] * inv;
        Aout[base + t + i * 256] = a;
        Abf [base + t + i * 256] = f2bf(a);
    }
}

// h = Qc + X ; LayerNorm over D ; write bf16 H
__global__ __launch_bounds__(256) void layernorm_kernel(
    const float* __restrict__ Qc, const float* __restrict__ Xf,
    const float* __restrict__ gamma, const float* __restrict__ beta,
    unsigned short* __restrict__ Hbf) {
    __shared__ float red[256];
    int row = blockIdx.x;               // 0..8191
    int t = threadIdx.x;
    size_t base = (size_t)row * D_DIM;
    float h[4]; float s = 0.f;
    #pragma unroll
    for (int j = 0; j < 4; ++j) {
        int idx = t + j * 256;
        h[j] = Qc[base + idx] + Xf[base + idx];
        s += h[j];
    }
    red[t] = s; __syncthreads();
    for (int st = 128; st > 0; st >>= 1) { if (t < st) red[t] += red[t + st]; __syncthreads(); }
    float mu = red[0] * (1.0f / D_DIM); __syncthreads();
    float vs = 0.f;
    #pragma unroll
    for (int j = 0; j < 4; ++j) { float d = h[j] - mu; vs += d * d; }
    red[t] = vs; __syncthreads();
    for (int st = 128; st > 0; st >>= 1) { if (t < st) red[t] += red[t + st]; __syncthreads(); }
    float rstd = rsqrtf(red[0] * (1.0f / D_DIM) + LN_EPS);
    #pragma unroll
    for (int j = 0; j < 4; ++j) {
        int idx = t + j * 256;
        float o = (h[j] - mu) * rstd * gamma[idx] + beta[idx];
        Hbf[base + idx] = f2bf(o);
    }
}

// ---------------- generic bf16 WMMA GEMM -----------------
// C[M,Nn] = A[M,K] * B      A row-major (lda == K)
// BT=false : B is Bsrc[K,Nn] row-major
// BT=true  : B is Bsrc[Nn,K] row-major (computes A * Bsrc^T); async double-buffered
// EPI: 0 = store bf16 ; 1 = logits (+bias,+mask) f32 ; 2 = guarded f32 store
#define BM 128
#define BN 64
#define BK 32
#define LDS_AS (BK + 8)   // 40 bf16 / row
#define LDS_BS (BK + 8)   // 40 bf16 / row ([n][k] layout)

template<int EPI, bool BT>
__global__ __launch_bounds__(256) void gemm_bf16_wmma_kernel(
    const unsigned short* __restrict__ Ag,
    const unsigned short* __restrict__ Bg,
    void* __restrict__ Cg,
    const int* __restrict__ Ms,
    int M, int Nn, int K,
    long long strideA, long long strideB, long long strideC)
{
    __shared__ unsigned short sA[2][BM * LDS_AS];
    __shared__ unsigned short sB[2][BN * LDS_BS];

    const int z = blockIdx.z;
    const unsigned short* A  = Ag + (size_t)z * strideA;
    const unsigned short* Bs = Bg + (size_t)z * strideB;

    const int m0  = blockIdx.y * BM;
    const int n0  = blockIdx.x * BN;
    const int tid = threadIdx.x;
    const int lane = tid & 31;
    const int wid  = tid >> 5;          // 8 waves
    const int wrow = (wid & 3) * 32;    // wave tile row in block tile
    const int wcol = (wid >> 2) * 32;   // wave tile col in block tile

    v8f zero8 = {0.f,0.f,0.f,0.f,0.f,0.f,0.f,0.f};
    v8f c[2][2];
    c[0][0] = zero8; c[0][1] = zero8; c[1][0] = zero8; c[1][1] = zero8;

    const int mr = lane & 15;
    const int g  = lane >> 4;

    auto compute_buf = [&](int buf) {
        Frag16 af[2], bfr[2];
        #pragma unroll
        for (int mi = 0; mi < 2; ++mi) {
            const unsigned short* p = &sA[buf][(wrow + mi * 16 + mr) * LDS_AS + g * 8];
            af[mi].q[0] = *(const uint4*)(p);        // k = g*8 .. g*8+7
            af[mi].q[1] = *(const uint4*)(p + 16);   // k = 16+g*8 ..
        }
        #pragma unroll
        for (int ni = 0; ni < 2; ++ni) {
            const unsigned short* p = &sB[buf][(wcol + ni * 16 + mr) * LDS_BS + g * 8];
            bfr[ni].q[0] = *(const uint4*)(p);       // k = g*8 .. g*8+7
            bfr[ni].q[1] = *(const uint4*)(p + 16);  // k = 16+g*8 ..
        }
        #pragma unroll
        for (int mi = 0; mi < 2; ++mi)
            #pragma unroll
            for (int ni = 0; ni < 2; ++ni)
                c[mi][ni] = __builtin_amdgcn_wmma_f32_16x16x32_bf16(
                    false, af[mi].v, false, bfr[ni].v,
                    (short)0, c[mi][ni], false, false);
    };

    if (BT) {
#ifdef HAS_ASYNC
        // ---------- async double-buffered pipeline (3 async ops / thread / tile) ----------
        const int ar = tid >> 2;              // 0..63
        const int ac = (tid & 3) * 8;         // 0,8,16,24
        int bn = n0 + ar;                     // B row (clamped; OOB cols are store-guarded)
        if (bn >= Nn) bn = Nn - 1;

        auto stage = [&](int k0, int buf) {
            async_copy16(A + (size_t)(m0 + ar)      * K + k0 + ac, &sA[buf][(ar)      * LDS_AS + ac]);
            async_copy16(A + (size_t)(m0 + ar + 64) * K + k0 + ac, &sA[buf][(ar + 64) * LDS_AS + ac]);
            async_copy16(Bs + (size_t)bn * K + k0 + ac,            &sB[buf][(ar)      * LDS_BS + ac]);
        };

        const int T = K / BK;
        stage(0, 0);
        for (int t = 0; t < T; ++t) {
            int cur = t & 1;
            if (t + 1 < T) {
                stage((t + 1) * BK, cur ^ 1);
                wait_async<3>();              // tile t's 3 ops retired; t+1 in flight
            } else {
                wait_async<0>();
            }
            __syncthreads();
            compute_buf(cur);
            __syncthreads();
        }
#else
        // ---------- synchronous fallback ----------
        for (int k0 = 0; k0 < K; k0 += BK) {
            int ar = tid >> 2, ac = (tid & 3) * 8;
            *(uint4*)(&sA[0][(ar)      * LDS_AS + ac]) =
                *(const uint4*)(A + (size_t)(m0 + ar)      * K + k0 + ac);
            *(uint4*)(&sA[0][(ar + 64) * LDS_AS + ac]) =
                *(const uint4*)(A + (size_t)(m0 + ar + 64) * K + k0 + ac);
            uint4 val = make_uint4(0u, 0u, 0u, 0u);
            if (n0 + ar < Nn)
                val = *(const uint4*)(Bs + (size_t)(n0 + ar) * K + k0 + ac);
            *(uint4*)(&sB[0][ar * LDS_BS + ac]) = val;
            __syncthreads();
            compute_buf(0);
            __syncthreads();
        }
#endif
    } else {
        // ---------- NN: async A staging, sync transposed B staging ----------
        for (int k0 = 0; k0 < BK * (K / BK); k0 += BK) {
            int ar = tid >> 2, ac = (tid & 3) * 8;
#ifdef HAS_ASYNC
            async_copy16(A + (size_t)(m0 + ar)      * K + k0 + ac, &sA[0][(ar)      * LDS_AS + ac]);
            async_copy16(A + (size_t)(m0 + ar + 64) * K + k0 + ac, &sA[0][(ar + 64) * LDS_AS + ac]);
#else
            *(uint4*)(&sA[0][(ar)      * LDS_AS + ac]) =
                *(const uint4*)(A + (size_t)(m0 + ar)      * K + k0 + ac);
            *(uint4*)(&sA[0][(ar + 64) * LDS_AS + ac]) =
                *(const uint4*)(A + (size_t)(m0 + ar + 64) * K + k0 + ac);
#endif
            {
                int bk  = tid >> 3;               // 0..31
                int bnc = (tid & 7) * 8;          // 0..56
                uint4 val = make_uint4(0u, 0u, 0u, 0u);
                if (n0 + bnc + 8 <= Nn)
                    val = *(const uint4*)(Bs + (size_t)(k0 + bk) * Nn + n0 + bnc);
                const unsigned short* pv = (const unsigned short*)&val;
                #pragma unroll
                for (int j = 0; j < 8; ++j)
                    sB[0][(bnc + j) * LDS_BS + bk] = pv[j];
            }
            if (k0 + BK < K)
                __builtin_prefetch((const void*)(A + (size_t)(m0 + (tid >> 2)) * K + k0 + BK), 0, 3);
            wait_async<0>();
            __syncthreads();
            compute_buf(0);
            __syncthreads();
        }
    }

    // ---- epilogue ----
    const int hi = lane >> 4;
    const int nl = lane & 15;
    #pragma unroll
    for (int mi = 0; mi < 2; ++mi) {
        #pragma unroll
        for (int ni = 0; ni < 2; ++ni) {
            #pragma unroll
            for (int r = 0; r < 8; ++r) {
                int m = m0 + wrow + mi * 16 + r + 8 * hi;
                int n = n0 + wcol + ni * 16 + nl;
                if (n >= Nn) continue;
                float val = c[mi][ni][r];
                if (EPI == 0) {
                    unsigned short* Cp = (unsigned short*)Cg + (size_t)z * strideC;
                    Cp[(size_t)m * Nn + n] = f2bf(val);
                } else if (EPI == 1) {
                    // logits: add ALiBi-style bias, apply pad mask
                    float* Cp = (float*)Cg + (size_t)z * strideC;
                    float bias = -fminf(fabsf((float)(n - m)), 5.0f);
                    int keytok = Ms[z * N_CTX + n];
                    val = (keytok != 0) ? (val + bias) : NEG_INF;
                    Cp[(size_t)m * Nn + n] = val;
                } else {
                    float* Cp = (float*)Cg + (size_t)z * strideC;
                    Cp[(size_t)m * Nn + n] = val;
                }
            }
        }
    }
}

// ---------------- host launcher ----------------
extern "C" void kernel_launch(void* const* d_in, const int* in_sizes, int n_in,
                              void* d_out, int out_size, void* d_ws, size_t ws_size,
                              hipStream_t stream) {
    (void)in_sizes; (void)n_in; (void)out_size; (void)ws_size;

    const int*   Ms    = (const int*)  d_in[0];   // [B,N]
    const float* Ev    = (const float*)d_in[1];   // [V,D]
    const float* Ey    = (const float*)d_in[2];   // [Y,D]
    const float* Wa    = (const float*)d_in[3];   // [D,D]
    const float* Wo    = (const float*)d_in[4];   // [D,D]
    const float* gamma = (const float*)d_in[5];   // [D]
    const float* beta  = (const float*)d_in[6];   // [D]

    float* y_out = (float*)d_out;                               // [B,N,Y]
    float* A_out = (float*)d_out + (size_t)ROWS * Y_VOC;        // [B,N,N]

    // workspace layout (bytes)
    char* ws = (char*)d_ws;
    size_t o = 0;
    unsigned short* Xbf  = (unsigned short*)(ws + o); o += (size_t)ROWS * D_DIM * 2;       // 16 MB
    unsigned short* WAbf = (unsigned short*)(ws + o); o += (size_t)D_DIM * D_DIM * 2;      //  2 MB
    unsigned short* WObf = (unsigned short*)(ws + o); o += (size_t)D_DIM * D_DIM * 2;      //  2 MB
    unsigned short* EYbf = (unsigned short*)(ws + o); o += (size_t)Y_VOC * D_DIM * 2;      // 40 MB
    unsigned short* QWbf = (unsigned short*)(ws + o); o += (size_t)ROWS * D_DIM * 2;       // 16 MB
    unsigned short* XObf = (unsigned short*)(ws + o); o += (size_t)ROWS * D_DIM * 2;       // 16 MB
    unsigned short* Abf  = (unsigned short*)(ws + o); o += (size_t)ROWS * N_CTX * 2;       // 32 MB
    unsigned short* Hbf  = (unsigned short*)(ws + o); o += (size_t)ROWS * D_DIM * 2;       // 16 MB
    float*          Xf   = (float*)(ws + o);          o += (size_t)ROWS * D_DIM * 4;       // 32 MB
    float*          Lg   = (float*)(ws + o);          o += (size_t)ROWS * N_CTX * 4;       // 64 MB
    float*          Qc   = (float*)(ws + o);          o += (size_t)ROWS * D_DIM * 4;       // 32 MB

    // 1) weight / embedding conversions to bf16
    f32_to_bf16_kernel<<<2048, 256, 0, stream>>>(Wa, WAbf, (size_t)D_DIM * D_DIM);
    f32_to_bf16_kernel<<<2048, 256, 0, stream>>>(Wo, WObf, (size_t)D_DIM * D_DIM);
    f32_to_bf16_kernel<<<20000, 256, 0, stream>>>(Ey, EYbf, (size_t)Y_VOC * D_DIM);

    // 2) X = E_v[M_s]  (f32 + bf16 copies)
    gather_embed_kernel<<<ROWS, 256, 0, stream>>>(Ms, Ev, Xf, Xbf);

    // 3) QW = X @ W_A ; XO = X @ W_O   (M=8192, N=1024, K=1024)
    {
        dim3 grid(D_DIM / BN, ROWS / BM, 1);
        gemm_bf16_wmma_kernel<0, false><<<grid, 256, 0, stream>>>(
            Xbf, WAbf, QWbf, nullptr, ROWS, D_DIM, D_DIM, 0, 0, 0);
        gemm_bf16_wmma_kernel<0, false><<<grid, 256, 0, stream>>>(
            Xbf, WObf, XObf, nullptr, ROWS, D_DIM, D_DIM, 0, 0, 0);
    }

    // 4) logits = QW @ X^T + bias, mask   (per batch: M=N=2048, K=1024)
    {
        dim3 grid(N_CTX / BN, N_CTX / BM, B_DIM);
        gemm_bf16_wmma_kernel<1, true><<<grid, 256, 0, stream>>>(
            QWbf, Xbf, Lg, Ms, N_CTX, N_CTX, D_DIM,
            (long long)N_CTX * D_DIM, (long long)N_CTX * D_DIM,
            (long long)N_CTX * N_CTX);
    }

    // 5) softmax -> A (f32 to d_out) + Abf
    softmax_kernel<<<ROWS, 256, 0, stream>>>(Lg, A_out, Abf);

    // 6) Qc = A @ XO   (per batch: M=2048, N=1024, K=2048)
    {
        dim3 grid(D_DIM / BN, N_CTX / BM, B_DIM);
        gemm_bf16_wmma_kernel<2, false><<<grid, 256, 0, stream>>>(
            Abf, XObf, Qc, nullptr, N_CTX, D_DIM, N_CTX,
            (long long)N_CTX * N_CTX, (long long)N_CTX * D_DIM,
            (long long)N_CTX * D_DIM);
    }

    // 7) H = LN(Qc + X) -> bf16
    layernorm_kernel<<<ROWS, 256, 0, stream>>>(Qc, Xf, gamma, beta, Hbf);

    // 8) y_logits = H @ E_y^T   (M=8192, N=20000 ragged, K=1024)
    {
        dim3 grid((Y_VOC + BN - 1) / BN, ROWS / BM, 1);
        gemm_bf16_wmma_kernel<2, true><<<grid, 256, 0, stream>>>(
            Hbf, EYbf, y_out, nullptr, ROWS, Y_VOC, D_DIM, 0, 0, 0);
    }
}
